// Encoder_41437844472389
// MI455X (gfx1250) — matrix operand order repaired and checked by reading, not compiled, not run on hardware
//
#include <hip/hip_runtime.h>

// Problem constants (from reference)
#define B_    32
#define T_    512
#define E_    300
#define H_    512
#define FOURH 2048
#define KTOT  832   // 512 (U rows) + 300 (W rows) padded to 832 = 26*32
#define NKK   26    // KTOT / 32
#define NKK_H 16    // h region k-steps (512/32)
#define NKK_E 10    // emb region k-steps (320/32)

typedef __attribute__((ext_vector_type(16))) __bf16 v16bf;
typedef __attribute__((ext_vector_type(8)))  float  v8f;

// Byte strides inside the pre-swizzled buffers (fragment = 32 lanes * 32 B = 1024 B)
#define FRAG_BYTES   1024u
#define B_GATE_BYTES (32u * NKK * FRAG_BYTES)   // 851,968  (32 col-tiles per gate)
#define B_NG_BYTES   (NKK * FRAG_BYTES)         // 26,624
#define AH_MT_BYTES  (NKK_H * FRAG_BYTES)       // 16,384
#define AE_T_BYTES   (NKK_E * FRAG_BYTES)       // 10,240

// Workspace byte offsets
#define WS_C     0u          // c      : 32*512 f32      = 64 KB
#define WS_HF32  65536u      // h f32  : 32*512 f32      = 64 KB (final output copy)
#define WS_AH0   131072u     // Ah buf0: 2*16*32*16 bf16 = 32 KB (h in A-fragment order)
#define WS_AH1   163840u     // Ah buf1: 32 KB
#define WS_AEMB  196608u     // Aemb   : 2*512*10*32*16 bf16 = 10 MB
#define WS_BSW   10682368u   // Bsw    : 128*26*32*16 bf16   = 3.25 MB

// ---------------- helpers ----------------
__device__ __forceinline__ float sigmoid_f(float x) {
  return 1.0f / (1.0f + __expf(-x));
}
__device__ __forceinline__ float tanh_f(float x) {
  float e = __expf(2.0f * x);
  return (e - 1.0f) / (e + 1.0f);
}
__device__ __forceinline__ v16bf ldfrag(const char* p) {
  return *(const v16bf*)p;
}

// ---------------- zero init (c, hf32, Ah0, Ah1: first 192 KB of ws) --------
__global__ __launch_bounds__(256) void init_zero(float* p, int n) {
  int i = blockIdx.x * 256 + threadIdx.x;
  if (i < n) p[i] = 0.0f;
}

// ---------------- pre-swizzle combined [U;W;0] into WMMA B-fragment order --
// Bsw layout: [nt(128)][kk(26)][lane(32)][j(16)] bf16, with
//   k   = kk*32 + (lane>>4)*8 + (j<8 ? j : j+8)
//   col = nt*16 + (lane&15)          (ISA 7.12.2 16-bit B-matrix layout)
__global__ __launch_bounds__(256) void prep_B(const float* __restrict__ U,
                                              const float* __restrict__ W,
                                              __bf16* __restrict__ Bsw) {
  int idx = blockIdx.x * 256 + threadIdx.x;
  const int total = 128 * NKK * 32 * 16;   // 1,703,936
  if (idx >= total) return;
  int nt   = idx / (NKK * 512);
  int rem  = idx % (NKK * 512);
  int kk   = rem / 512;
  int rem2 = rem % 512;
  int lane = rem2 >> 4;
  int j    = rem2 & 15;
  int k    = kk * 32 + ((lane >> 4) << 3) + ((j < 8) ? j : j + 8);
  int col  = (nt << 4) + (lane & 15);
  float v = 0.0f;
  if (k < H_)            v = U[k * FOURH + col];
  else if (k < H_ + E_)  v = W[(k - H_) * FOURH + col];
  Bsw[idx] = (__bf16)v;
}

// ---------------- pre-gather embeddings into WMMA A-fragment order ---------
// Aemb layout: [mtile(2)][t(512)][kk(10)][lane(32)][j(16)] bf16, with
//   m = mtile*16 + (lane&15)
//   e = kk*32 + (lane>>4)*8 + (j<8 ? j : j+8)   (ISA 16-bit A layout)
//   value = e < 300 ? emb[tokens[m][t]][e] : 0
__global__ __launch_bounds__(256) void prep_Aemb(const int*   __restrict__ tokens,
                                                 const float* __restrict__ emb,
                                                 __bf16* __restrict__ Aemb) {
  int idx = blockIdx.x * 256 + threadIdx.x;
  const int total = 2 * T_ * NKK_E * 32 * 16;   // 5,242,880
  if (idx >= total) return;
  int mtile = idx / (T_ * NKK_E * 512);
  int r1    = idx % (T_ * NKK_E * 512);
  int t     = r1 / (NKK_E * 512);
  int r2    = r1 % (NKK_E * 512);
  int kk    = r2 / 512;
  int r3    = r2 % 512;
  int lane  = r3 >> 4;
  int j     = r3 & 15;
  int khalf = lane >> 4;
  int e     = kk * 32 + khalf * 8 + ((j < 8) ? j : j + 8);
  int m     = (mtile << 4) + (lane & 15);
  float v = 0.0f;
  if (e < E_) {
    int tok = tokens[m * T_ + t];
    v = emb[(long)tok * E_ + e];
  }
  Aemb[idx] = (__bf16)v;
}

// ---------------- one LSTM timestep ----------------
// grid = (2 m-tiles, 32 n-groups), block = 32 threads (1 wave).
// All fragment addresses are a loop-invariant base + compile-time immediate
// byte offsets (the kk/gate loops are fully unrolled); per k-step the 4 gate
// B-fragments load into independent registers so the 8 b128 loads clause
// together and the 4 wmmas issue back-to-back.
__global__ __launch_bounds__(32) void lstm_step(const float*  __restrict__ bias,
                                                const __bf16* __restrict__ Bsw,
                                                const __bf16* __restrict__ Aemb,
                                                const __bf16* __restrict__ Ah_in,
                                                __bf16*       __restrict__ Ah_out,
                                                float*        __restrict__ c,
                                                float*        __restrict__ hf32,
                                                int t) {
  const int lane  = threadIdx.x;        // 0..31
  const int mtile = blockIdx.x;         // 0..1
  const int ng    = blockIdx.y;         // 0..31
  const int nloc  = lane & 15;
  const int khalf = lane >> 4;
  const int ncol  = (ng << 4) + nloc;   // column within H for this lane
  const size_t laneB = (size_t)lane * 32u;

  // Loop-invariant byte base pointers; all loop offsets become immediates.
  const char* bbase = (const char*)Bsw + (size_t)ng * B_NG_BYTES + laneB;
  const char* abh   = (const char*)Ah_in + (size_t)mtile * AH_MT_BYTES + laneB;
  const char* abe   = (const char*)Aemb +
                      (size_t)(mtile * T_ + t) * AE_T_BYTES + laneB;

  // Accumulators initialized with bias (bias depends only on column).
  v8f acc[4];
#pragma unroll
  for (int g = 0; g < 4; ++g) {
    float bv = bias[g * H_ + ncol];
#pragma unroll
    for (int r = 0; r < 8; ++r) acc[g][r] = bv;
  }

#pragma unroll
  for (int kk = 0; kk < NKK; ++kk) {
    v16bf a = (kk < NKK_H) ? ldfrag(abh + (unsigned)kk * FRAG_BYTES)
                           : ldfrag(abe + (unsigned)(kk - NKK_H) * FRAG_BYTES);
    const char* bk = bbase + (unsigned)kk * FRAG_BYTES;
    v16bf b0 = ldfrag(bk);
    v16bf b1 = ldfrag(bk + 1u * B_GATE_BYTES);
    v16bf b2 = ldfrag(bk + 2u * B_GATE_BYTES);
    v16bf b3 = ldfrag(bk + 3u * B_GATE_BYTES);
    acc[0] = __builtin_amdgcn_wmma_f32_16x16x32_bf16(false, a, false, b0,
                                                     (short)0, acc[0], false, false);
    acc[1] = __builtin_amdgcn_wmma_f32_16x16x32_bf16(false, a, false, b1,
                                                     (short)0, acc[1], false, false);
    acc[2] = __builtin_amdgcn_wmma_f32_16x16x32_bf16(false, a, false, b2,
                                                     (short)0, acc[2], false, false);
    acc[3] = __builtin_amdgcn_wmma_f32_16x16x32_bf16(false, a, false, b3,
                                                     (short)0, acc[3], false, false);
  }

  // Gate math + state update.
  // C/D layout: VGPR r -> M = r + khalf*8 ; N = lane & 15.
  // Scatter coords for next step's A fragment (fixed per lane: k = ncol):
  const int kk_d   = ncol >> 5;
  const int klocal = ncol & 31;
  const int rdhalf = (klocal >> 3) & 1;
  const int j_d    = (klocal & 7) | ((klocal >> 4) << 3);
#pragma unroll
  for (int r = 0; r < 8; ++r) {
    const int mloc = khalf * 8 + r;              // 0..15
    const int m    = (mtile << 4) + mloc;
    const int idx  = m * H_ + ncol;
    float i = sigmoid_f(acc[0][r]);
    float f = sigmoid_f(acc[1][r]);
    float g = tanh_f(acc[2][r]);
    float o = sigmoid_f(acc[3][r]);
    float cn = f * c[idx] + i * g;
    float hn = o * tanh_f(cn);
    c[idx]    = cn;
    hf32[idx] = hn;
    const int rd_lane = rdhalf * 16 + mloc;
    Ah_out[(((mtile * NKK_H + kk_d) * 32) + rd_lane) * 16 + j_d] = (__bf16)hn;
  }
}

// ---------------- final output: [hidden | cell] ----------------
__global__ __launch_bounds__(256) void write_out(const float* __restrict__ h,
                                                 const float* __restrict__ c,
                                                 float* __restrict__ out) {
  int i = blockIdx.x * 256 + threadIdx.x;
  if (i < B_ * H_) {
    out[i]           = h[i];
    out[B_ * H_ + i] = c[i];
  }
}

extern "C" void kernel_launch(void* const* d_in, const int* in_sizes, int n_in,
                              void* d_out, int out_size, void* d_ws, size_t ws_size,
                              hipStream_t stream) {
  (void)in_sizes; (void)n_in; (void)out_size; (void)ws_size;
  const int*   tokens = (const int*)d_in[0];     // [32, 512]
  const float* emb    = (const float*)d_in[1];   // [20000, 300]
  const float* W      = (const float*)d_in[2];   // [300, 2048]
  const float* U      = (const float*)d_in[3];   // [512, 2048]
  const float* bias   = (const float*)d_in[4];   // [2048]
  float* out = (float*)d_out;                    // [2, 32, 512]

  char* ws = (char*)d_ws;
  float*  c    = (float*)(ws + WS_C);
  float*  hf32 = (float*)(ws + WS_HF32);
  __bf16* Ah0  = (__bf16*)(ws + WS_AH0);
  __bf16* Ah1  = (__bf16*)(ws + WS_AH1);
  __bf16* Aemb = (__bf16*)(ws + WS_AEMB);
  __bf16* Bsw  = (__bf16*)(ws + WS_BSW);

  // 1) zero c, hf32, Ah0, Ah1 (contiguous 192 KB = 49152 floats)
  {
    int n = 49152;
    init_zero<<<(n + 255) / 256, 256, 0, stream>>>((float*)ws, n);
  }

  // 2) pre-swizzle [U;W;0] into WMMA B-fragment order (bf16)
  {
    int total = 128 * NKK * 32 * 16;
    prep_B<<<(total + 255) / 256, 256, 0, stream>>>(U, W, Bsw);
  }

  // 3) pre-gather embedding A-fragments for all timesteps (bf16)
  {
    int total = 2 * T_ * NKK_E * 32 * 16;
    prep_Aemb<<<(total + 255) / 256, 256, 0, stream>>>(tokens, emb, Aemb);
  }

  // 4) 512 sequential LSTM steps (Ah double-buffered, c in place)
  dim3 grid(2, 32);
  for (int t = 0; t < T_; ++t) {
    const __bf16* a_in  = (t & 1) ? Ah1 : Ah0;
    __bf16*       a_out = (t & 1) ? Ah0 : Ah1;
    lstm_step<<<grid, 32, 0, stream>>>(bias, Bsw, Aemb, a_in, a_out, c, hf32, t);
  }

  // 5) emit (hidden, cell)
  {
    int n = B_ * H_;
    write_out<<<(n + 255) / 256, 256, 0, stream>>>(hf32, c, out);
  }
}
